// AttentionSeq2Seq_79379585564750
// MI455X (gfx1250) — matrix operand order repaired and checked by reading, not compiled, not run on hardware
//
#include <hip/hip_runtime.h>
#include <hip/hip_bf16.h>
#include <math.h>

// ---------------- problem constants (match reference) ----------------
#define SRC_V 16000
#define TRG_V 32000
#define E_DIM 256
#define H_DIM 512
#define B_DIM 32
#define S_DIM 64
#define T_DIM 64
#define G_DIM (4 * H_DIM)          // 2048 gate width
#define KCAT  (E_DIM + 2 * H_DIM)  // 1280 concat width for decoder / FC

typedef __attribute__((ext_vector_type(16))) _Float16 v16h;
typedef __attribute__((ext_vector_type(8)))  float    v8f;
typedef __attribute__((ext_vector_type(4)))  float    f32x4;

#define WAVES_PER_BLOCK 8

// =====================================================================
// Generic WMMA GEMM:  out[M,N] = A[M,K] * W^T  (+bias0 +bias1 +addin)
// A row-major f16, W row-major [N, K] f16 (optionally split in K into
// W0 [N,Ksplit] and W1 [N,K-Ksplit]). Each wave computes one 16x16 tile
// with v_wmma_f32_16x16x32_f16.
// =====================================================================
__global__ __launch_bounds__(256) void wmma_gemm_f16(
    const _Float16* __restrict__ A,
    const _Float16* __restrict__ W0,
    const _Float16* __restrict__ W1,
    float* __restrict__ outp, long long ldo,
    const float* __restrict__ bias0,
    const float* __restrict__ bias1,
    const float* __restrict__ addin,   // [M,N] row-major or null
    int M, int N, int K, int Ksplit)
{
  const int lane = threadIdx.x & 31;
  const int wav  = threadIdx.x >> 5;
  const int half = lane >> 4;    // 0: lanes 0-15, 1: lanes 16-31
  const int l16  = lane & 15;
  const int Mt = M >> 4, Nt = N >> 4;
  const int K1 = K - Ksplit;

  for (int tile = blockIdx.x * WAVES_PER_BLOCK + wav; tile < Mt * Nt;
       tile += gridDim.x * WAVES_PER_BLOCK) {
    const int mt = tile % Mt;
    const int nt = tile / Mt;
    const int row_a = (mt << 4) + l16;   // A row this lane feeds (M index)
    const int col_n = (nt << 4) + l16;   // output column this lane owns
    const _Float16* arow = A + (size_t)row_a * K;

    v8f acc = {0.f, 0.f, 0.f, 0.f, 0.f, 0.f, 0.f, 0.f};

    for (int kk = 0; kk < K; kk += 32) {
      // ---- A operand: 16x32 f16 tile, ISA layout:
      // lane(0-15)=row M, elems 0..7 -> K=kk+half*8+e, elems 8..15 -> +16
      union { v16h v; f32x4 f[2]; } au, bu;
      const _Float16* ap = arow + kk + (half << 3);
      au.f[0] = *(const f32x4*)(ap);
      au.f[1] = *(const f32x4*)(ap + 16);

      // ---- B operand: 32x16 f16 tile (W^T). lane owns column N=col_n,
      // elems 0..7 -> K = kk + half*8 + e, elems 8..15 -> +16 (matches A).
      const _Float16* bp;
      if (kk < Ksplit) bp = W0 + (size_t)col_n * Ksplit + kk;
      else             bp = W1 + (size_t)col_n * K1 + (kk - Ksplit);
      bp += (half << 3);
      bu.f[0] = *(const f32x4*)(bp);
      bu.f[1] = *(const f32x4*)(bp + 16);
      __builtin_prefetch(bp + 32, 0, 1);   // next K block -> global_prefetch

      acc = __builtin_amdgcn_wmma_f32_16x16x32_f16(
          /*neg_a=*/false, au.v, /*neg_b=*/false, bu.v,
          /*c_mod=*/(short)0, acc, /*reuse_a=*/false, /*reuse_b=*/false);
    }

    const int n = col_n;
    float badd = (bias0 ? bias0[n] : 0.f) + (bias1 ? bias1[n] : 0.f);
#pragma unroll
    for (int j = 0; j < 8; ++j) {
      // C/D layout: VGPR j -> M = mt*16 + half*8 + j ; N = lane&15
      int m = (mt << 4) + (half << 3) + j;
      float v = acc[j] + badd;
      if (addin) v += addin[(size_t)m * N + n];
      outp[(size_t)m * ldo + n] = v;
    }
  }
}

// =====================================================================
// Elementwise / utility kernels
// =====================================================================
__global__ void f32_to_f16_kernel(const float* __restrict__ s,
                                  _Float16* __restrict__ d, size_t n) {
  size_t i = (size_t)blockIdx.x * blockDim.x + threadIdx.x;
  size_t st = (size_t)gridDim.x * blockDim.x;
  for (; i < n; i += st) d[i] = (_Float16)s[i];
}

// attn_W [H, 2H] -> Wh [H,H] (cols 0..H-1), We [H,H] (cols H..2H-1), f16
__global__ void split_attn_kernel(const float* __restrict__ attn_W,
                                  _Float16* __restrict__ Wh,
                                  _Float16* __restrict__ We) {
  int i = blockIdx.x * blockDim.x + threadIdx.x;
  if (i >= H_DIM * H_DIM) return;
  int n = i / H_DIM, k = i % H_DIM;
  Wh[i] = (_Float16)attn_W[(size_t)n * 2 * H_DIM + k];
  We[i] = (_Float16)attn_W[(size_t)n * 2 * H_DIM + H_DIM + k];
}

// xe[s*B + b][e] = emb_src[src[b,s], e]   (time-major rows, f16)
__global__ void gather_src_kernel(const int* __restrict__ src,
                                  const float* __restrict__ emb_src,
                                  _Float16* __restrict__ xe) {
  int i = blockIdx.x * blockDim.x + threadIdx.x;
  if (i >= S_DIM * B_DIM * E_DIM) return;
  int s = i / (B_DIM * E_DIM);
  int r = i % (B_DIM * E_DIM);
  int b = r / E_DIM, e = r % E_DIM;
  int tok = src[b * S_DIM + s];
  xe[i] = (_Float16)emb_src[(size_t)tok * E_DIM + e];
}

__global__ void init_state_kernel(float* h, float* c, _Float16* h_h) {
  int i = blockIdx.x * blockDim.x + threadIdx.x;
  if (i < B_DIM * H_DIM) { h[i] = 0.f; c[i] = 0.f; h_h[i] = (_Float16)0.f; }
}

// zero outputs[:,0,:] and attention[:,0,:] (d_out is poisoned pre-timing)
__global__ void zero_t0_kernel(float* __restrict__ out) {
  size_t nv = (size_t)B_DIM * TRG_V;
  size_t tot = nv + (size_t)B_DIM * S_DIM;
  size_t i = (size_t)blockIdx.x * blockDim.x + threadIdx.x;
  size_t st = (size_t)gridDim.x * blockDim.x;
  for (; i < tot; i += st) {
    if (i < nv) {
      size_t b = i / TRG_V, v = i % TRG_V;
      out[b * (size_t)T_DIM * TRG_V + v] = 0.f;
    } else {
      size_t j = i - nv, b = j / S_DIM, s = j % S_DIM;
      out[(size_t)B_DIM * T_DIM * TRG_V + b * (size_t)T_DIM * S_DIM + s] = 0.f;
    }
  }
}

// LSTM cell (PyTorch gate order i,f,g,o). Optionally records encoder
// output at step t, optionally writes new h into ycat[:, 0:H] (decoder).
__global__ void lstm_cell_kernel(const float* __restrict__ g,
                                 float* __restrict__ h, float* __restrict__ c,
                                 _Float16* __restrict__ h_h,
                                 float* __restrict__ enc_out,
                                 _Float16* __restrict__ enc_out_h, int t,
                                 _Float16* __restrict__ ycat) {
  int idx = blockIdx.x * blockDim.x + threadIdx.x;
  if (idx >= B_DIM * H_DIM) return;
  int b = idx >> 9, hh = idx & (H_DIM - 1);
  const float* gr = g + (size_t)b * G_DIM;
  float gi = gr[hh];
  float gf = gr[H_DIM + hh];
  float gg = gr[2 * H_DIM + hh];
  float go = gr[3 * H_DIM + hh];
  float iv = 1.f / (1.f + expf(-gi));
  float fv = 1.f / (1.f + expf(-gf));
  float ov = 1.f / (1.f + expf(-go));
  float cn = fv * c[idx] + iv * tanhf(gg);
  float hn = ov * tanhf(cn);
  c[idx] = cn;
  h[idx] = hn;
  h_h[idx] = (_Float16)hn;
  if (enc_out) {
    size_t eidx = ((size_t)b * S_DIM + t) * H_DIM + hh;
    enc_out[eidx] = hn;
    enc_out_h[eidx] = (_Float16)hn;
  }
  if (ycat) ycat[b * KCAT + hh] = (_Float16)hn;
}

// score[b,s] = sum_h v_w[h]*tanh(h_part[b,h] + enc_part[b*S+s,h]), masked.
// One wave per (b,s) row.
__global__ __launch_bounds__(256) void attn_score_kernel(
    const float* __restrict__ h_part, const float* __restrict__ enc_part,
    const float* __restrict__ v_w, const int* __restrict__ src,
    float* __restrict__ score) {
  int wav = (blockIdx.x * blockDim.x + threadIdx.x) >> 5;
  int lane = threadIdx.x & 31;
  if (wav >= B_DIM * S_DIM) return;
  int b = wav >> 6;  // S=64
  float sum = 0.f;
  const float* hp = h_part + (size_t)b * H_DIM;
  const float* ep = enc_part + (size_t)wav * H_DIM;
  for (int hh = lane; hh < H_DIM; hh += 32)
    sum += v_w[hh] * tanhf(hp[hh] + ep[hh]);
  for (int off = 16; off > 0; off >>= 1) sum += __shfl_xor(sum, off, 32);
  if (lane == 0) score[wav] = (src[wav] != 0) ? sum : -1e10f;
}

// Per-batch softmax over S, attention output, context vector, and builds
// f16 concat rows:  xcat=[emb|ctx|h_old] (LSTM in), ycat[ctx|emb parts].
__global__ __launch_bounds__(256) void softmax_ctx_kernel(
    const float* __restrict__ score, const float* __restrict__ enc_out,
    const float* __restrict__ emb_trg, const int* __restrict__ trg,
    const _Float16* __restrict__ h_h,
    float* __restrict__ attn_out,   // d_out attention section
    float* __restrict__ ctx,
    _Float16* __restrict__ xcat, _Float16* __restrict__ ycat, int t) {
  __shared__ float sa[S_DIM];
  __shared__ float sinv;
  int b = blockIdx.x, tid = threadIdx.x;
  if (tid < S_DIM) sa[tid] = score[b * S_DIM + tid];
  __syncthreads();
  if (tid == 0) {
    float mx = sa[0];
    for (int s = 1; s < S_DIM; ++s) mx = fmaxf(mx, sa[s]);
    float den = 0.f;
    for (int s = 0; s < S_DIM; ++s) { sa[s] = expf(sa[s] - mx); den += sa[s]; }
    sinv = 1.f / den;
  }
  __syncthreads();
  float inv = sinv;
  if (tid < S_DIM) {
    float a = sa[tid] * inv;
    sa[tid] = a;
    attn_out[(size_t)b * T_DIM * S_DIM + (size_t)t * S_DIM + tid] = a;
  }
  __syncthreads();
  // context = attn . enc_out   and f16 concat fills
  for (int hh = tid; hh < H_DIM; hh += 256) {
    float acc = 0.f;
    const float* eo = enc_out + ((size_t)b * S_DIM) * H_DIM + hh;
    for (int s = 0; s < S_DIM; ++s) acc += sa[s] * eo[(size_t)s * H_DIM];
    ctx[b * H_DIM + hh] = acc;
    xcat[b * KCAT + E_DIM + hh] = (_Float16)acc;          // [E, E+H)
    ycat[b * KCAT + H_DIM + hh] = (_Float16)acc;          // [H, 2H)
  }
  int tok = trg[b * T_DIM + (t - 1)];                     // teacher forcing
  for (int e = tid; e < E_DIM; e += 256) {
    float v = emb_trg[(size_t)tok * E_DIM + e];
    xcat[b * KCAT + e] = (_Float16)v;                     // [0, E)
    ycat[b * KCAT + 2 * H_DIM + e] = (_Float16)v;         // [2H, 2H+E)
  }
  for (int hh = tid; hh < H_DIM; hh += 256)
    xcat[b * KCAT + E_DIM + H_DIM + hh] = h_h[b * H_DIM + hh]; // [E+H, E+2H)
}

// =====================================================================
// Host orchestration
// =====================================================================
extern "C" void kernel_launch(void* const* d_in, const int* in_sizes, int n_in,
                              void* d_out, int out_size, void* d_ws, size_t ws_size,
                              hipStream_t stream) {
  const int*   src     = (const int*)d_in[0];
  const int*   trg     = (const int*)d_in[1];
  const float* emb_src = (const float*)d_in[2];
  const float* Wih_e   = (const float*)d_in[3];
  const float* Whh_e   = (const float*)d_in[4];
  const float* bih_e   = (const float*)d_in[5];
  const float* bhh_e   = (const float*)d_in[6];
  const float* emb_trg = (const float*)d_in[7];
  const float* attn_W  = (const float*)d_in[8];
  const float* attn_b  = (const float*)d_in[9];
  const float* v_w     = (const float*)d_in[10];
  const float* Wih_d   = (const float*)d_in[11];
  const float* Whh_d   = (const float*)d_in[12];
  const float* bih_d   = (const float*)d_in[13];
  const float* bhh_d   = (const float*)d_in[14];
  const float* fc_W    = (const float*)d_in[15];
  const float* fc_b    = (const float*)d_in[16];
  float* out = (float*)d_out;

  // ---- carve workspace (~120 MB), 256B-aligned chunks ----
  char* p = (char*)d_ws;
  auto alloc = [&](size_t bytes) -> void* {
    void* r = (void*)p; p += (bytes + 255) & ~(size_t)255; return r;
  };
  _Float16* WihE_h  = (_Float16*)alloc((size_t)G_DIM * E_DIM * 2);
  _Float16* WhhE_h  = (_Float16*)alloc((size_t)G_DIM * H_DIM * 2);
  _Float16* WihD_h  = (_Float16*)alloc((size_t)G_DIM * (E_DIM + H_DIM) * 2);
  _Float16* WhhD_h  = (_Float16*)alloc((size_t)G_DIM * H_DIM * 2);
  _Float16* Wh_h    = (_Float16*)alloc((size_t)H_DIM * H_DIM * 2);
  _Float16* We_h    = (_Float16*)alloc((size_t)H_DIM * H_DIM * 2);
  _Float16* fcW_h   = (_Float16*)alloc((size_t)TRG_V * KCAT * 2);     // 82 MB, L2-resident
  _Float16* xe_h    = (_Float16*)alloc((size_t)S_DIM * B_DIM * E_DIM * 2);
  float*    encX    = (float*)alloc((size_t)S_DIM * B_DIM * G_DIM * 4); // x@Wih_e^T all steps
  float*    g       = (float*)alloc((size_t)B_DIM * G_DIM * 4);
  float*    h       = (float*)alloc((size_t)B_DIM * H_DIM * 4);
  float*    c       = (float*)alloc((size_t)B_DIM * H_DIM * 4);
  _Float16* h_h     = (_Float16*)alloc((size_t)B_DIM * H_DIM * 2);
  float*    enc_out = (float*)alloc((size_t)B_DIM * S_DIM * H_DIM * 4);
  _Float16* enc_out_h = (_Float16*)alloc((size_t)B_DIM * S_DIM * H_DIM * 2);
  float*    enc_part  = (float*)alloc((size_t)B_DIM * S_DIM * H_DIM * 4);
  float*    h_part  = (float*)alloc((size_t)B_DIM * H_DIM * 4);
  float*    score   = (float*)alloc((size_t)B_DIM * S_DIM * 4);
  float*    ctx     = (float*)alloc((size_t)B_DIM * H_DIM * 4);
  _Float16* xcat    = (_Float16*)alloc((size_t)B_DIM * KCAT * 2);
  _Float16* ycat    = (_Float16*)alloc((size_t)B_DIM * KCAT * 2);

  auto gemm = [&](const _Float16* A, const _Float16* W0, const _Float16* W1,
                  float* o, long long ldo, const float* b0, const float* b1,
                  const float* add, int M, int N, int K, int Ksplit) {
    int tiles = (M / 16) * (N / 16);
    int blocks = (tiles + WAVES_PER_BLOCK - 1) / WAVES_PER_BLOCK;
    wmma_gemm_f16<<<blocks, 256, 0, stream>>>(A, W0, W1, o, ldo, b0, b1, add,
                                              M, N, K, Ksplit);
  };

  // ---- one-time prep ----
  zero_t0_kernel<<<2048, 256, 0, stream>>>(out);
  f32_to_f16_kernel<<<512, 256, 0, stream>>>(Wih_e, WihE_h, (size_t)G_DIM * E_DIM);
  f32_to_f16_kernel<<<512, 256, 0, stream>>>(Whh_e, WhhE_h, (size_t)G_DIM * H_DIM);
  f32_to_f16_kernel<<<512, 256, 0, stream>>>(Wih_d, WihD_h, (size_t)G_DIM * (E_DIM + H_DIM));
  f32_to_f16_kernel<<<512, 256, 0, stream>>>(Whh_d, WhhD_h, (size_t)G_DIM * H_DIM);
  f32_to_f16_kernel<<<4096, 256, 0, stream>>>(fc_W, fcW_h, (size_t)TRG_V * KCAT);
  split_attn_kernel<<<(H_DIM * H_DIM + 255) / 256, 256, 0, stream>>>(attn_W, Wh_h, We_h);
  gather_src_kernel<<<(S_DIM * B_DIM * E_DIM + 255) / 256, 256, 0, stream>>>(src, emb_src, xe_h);
  init_state_kernel<<<(B_DIM * H_DIM + 255) / 256, 256, 0, stream>>>(h, c, h_h);

  // encX[t*B+b, :] = x_emb @ Wih_e^T  (all timesteps, one big WMMA GEMM)
  gemm(xe_h, WihE_h, nullptr, encX, G_DIM, nullptr, nullptr, nullptr,
       S_DIM * B_DIM, G_DIM, E_DIM, E_DIM);

  // ---- encoder: 64 sequential LSTM steps ----
  for (int t = 0; t < S_DIM; ++t) {
    gemm(h_h, WhhE_h, nullptr, g, G_DIM, bih_e, bhh_e,
         encX + (size_t)t * B_DIM * G_DIM, B_DIM, G_DIM, H_DIM, H_DIM);
    lstm_cell_kernel<<<(B_DIM * H_DIM + 255) / 256, 256, 0, stream>>>(
        g, h, c, h_h, enc_out, enc_out_h, t, nullptr);
  }

  // enc_part = enc_out @ We^T + attn_b   (loop-invariant attention half)
  gemm(enc_out_h, We_h, nullptr, enc_part, H_DIM, attn_b, nullptr, nullptr,
       B_DIM * S_DIM, H_DIM, H_DIM, H_DIM);

  // ---- decoder: 63 sequential steps, teacher forcing ----
  for (int t = 1; t < T_DIM; ++t) {
    // h_part = h @ Wh^T
    gemm(h_h, Wh_h, nullptr, h_part, H_DIM, nullptr, nullptr, nullptr,
         B_DIM, H_DIM, H_DIM, H_DIM);
    attn_score_kernel<<<(B_DIM * S_DIM * 32 + 255) / 256, 256, 0, stream>>>(
        h_part, enc_part, v_w, src, score);
    softmax_ctx_kernel<<<B_DIM, 256, 0, stream>>>(
        score, enc_out, emb_trg, trg, h_h,
        out + (size_t)B_DIM * T_DIM * TRG_V, ctx, xcat, ycat, t);
    // gates = [emb|ctx] @ Wih_d^T + h @ Whh_d^T + biases  (K-split GEMM)
    gemm(xcat, WihD_h, WhhD_h, g, G_DIM, bih_d, bhh_d, nullptr,
         B_DIM, G_DIM, KCAT, E_DIM + H_DIM);
    lstm_cell_kernel<<<(B_DIM * H_DIM + 255) / 256, 256, 0, stream>>>(
        g, h, c, h_h, nullptr, nullptr, 0, ycat);
    // preds[:, t, :] = [h|ctx|emb] @ fc_W^T + fc_b  (dominant GEMM, f16 fc_W in L2)
    gemm(ycat, fcW_h, nullptr, out + (size_t)t * TRG_V, (long long)T_DIM * TRG_V,
         fc_b, nullptr, nullptr, B_DIM, TRG_V, KCAT, KCAT);
  }
}
